// CTCBridgeSparseSlot_63462436765728
// MI455X (gfx1250) — matrix operand order repaired and checked by reading, not compiled, not run on hardware
//
#include <hip/hip_runtime.h>
#include <hip/hip_bf16.h>
#include <math.h>

// ---------------------------------------------------------------------------
// CDNA5 (gfx1250): all matmuls via v_wmma_f32_16x16x32_bf16; flash-attention
// K/V tiles streamed by the Tensor Data Mover (tensor_load_to_lds +
// s_wait_tensorcnt) with TDM LDS padding giving a 72-half conflict-free pitch.
// ---------------------------------------------------------------------------

typedef __attribute__((ext_vector_type(16))) __bf16 v16bf;
typedef __attribute__((ext_vector_type(8)))  float  v8f;
typedef __attribute__((ext_vector_type(4)))  unsigned int u32x4;
typedef __attribute__((ext_vector_type(8)))  int i32x8;
typedef __attribute__((ext_vector_type(4)))  int i32x4;

union Frag16 {
  unsigned short s[16];
  v16bf b;
};

__device__ __forceinline__ unsigned short f2bf(float f) {
  unsigned int u = __float_as_uint(f);
  u += 0x7FFFu + ((u >> 16) & 1u);          // round-to-nearest-even
  return (unsigned short)(u >> 16);
}

__device__ __forceinline__ v8f v8f_zero() {
  v8f z;
#pragma unroll
  for (int i = 0; i < 8; ++i) z[i] = 0.0f;
  return z;
}

// ---- Tensor Data Mover availability ---------------------------------------
#if defined(__HIP_DEVICE_COMPILE__) && defined(__has_builtin)
#  if __has_builtin(__builtin_amdgcn_tensor_load_to_lds) && \
      __has_builtin(__builtin_amdgcn_s_wait_tensorcnt)
#    define HAVE_TDM 1
#  endif
#endif
#ifndef HAVE_TDM
#define HAVE_TDM 0
#endif

#if HAVE_TDM
// 2D bf16 tile load: tile_w elems/row, tile_h rows, row stride in elems.
// LDS pad: +4 dwords after every 32 dwords -> effective pitch 72 halves.
__device__ __forceinline__ void tdm_load_2d_bf16(unsigned lds_byte_off,
                                                 const unsigned short* gptr,
                                                 unsigned tile_w, unsigned tile_h,
                                                 unsigned row_stride_elems) {
  unsigned long long ga = (unsigned long long)(uintptr_t)gptr;
  u32x4 g0;
  g0[0] = 1u;                                       // count=1 (valid D#)
  g0[1] = lds_byte_off;                             // LDS dest (bytes)
  g0[2] = (unsigned)ga;                             // global_addr[31:0]
  g0[3] = (unsigned)((ga >> 32) & 0x01FFFFFFull) | (2u << 30);  // [56:32]|type=2
  i32x8 g1;
  // wg_mask=0 | data_size=1(2B) | pad_enable | pad_interval=4(32dw) | pad_amount=3(4dw)
  g1[0] = (int)((1u << 16) | (1u << 20) | (4u << 22) | (3u << 25));
  g1[1] = (int)(tile_w << 16);                      // tensor_dim0[15:0]
  g1[2] = (int)(tile_h << 16);                      // tensor_dim1[15:0]
  g1[3] = (int)(tile_w << 16);                      // tile_dim0
  g1[4] = (int)tile_h;                              // tile_dim1 (tile_dim2=0)
  g1[5] = (int)row_stride_elems;                    // tensor_dim0_stride[31:0]
  g1[6] = 0;
  g1[7] = 0;
  i32x4 z4; z4[0] = 0; z4[1] = 0; z4[2] = 0; z4[3] = 0;
#if __clang_major__ >= 23
  i32x8 z8;
  for (int i = 0; i < 8; ++i) z8[i] = 0;
  __builtin_amdgcn_tensor_load_to_lds(g0, g1, z4, z4, z8, 0);
#else
  __builtin_amdgcn_tensor_load_to_lds(g0, g1, z4, z4, 0);
#endif
}
#endif

// ---------------------------------------------------------------------------
// Weight pack: 10 x 512x512 bf16 blocks:
// 0:W_mem 1:Wkh 2:Wvh 3:Wqh 4:W_q 5..7:W_kv[k][:, :512] 8:W_attn_out 9:W_o
// ---------------------------------------------------------------------------
__global__ void pack_weights_kernel(const float* __restrict__ Wmem,
                                    const float* __restrict__ Wqkv,
                                    const float* __restrict__ Wq,
                                    const float* __restrict__ Wkv,
                                    const float* __restrict__ Wao,
                                    const float* __restrict__ Wo,
                                    unsigned short* __restrict__ dst) {
  int i = blockIdx.x * blockDim.x + threadIdx.x;
  if (i >= 512 * 512) return;
  int r = i >> 9, c = i & 511;
  const int WSZ = 512 * 512;
  dst[0 * WSZ + i] = f2bf(Wmem[i]);
  dst[1 * WSZ + i] = f2bf(Wqkv[r * 1536 + 512 + c]);    // Wkh
  dst[2 * WSZ + i] = f2bf(Wqkv[r * 1536 + 1024 + c]);   // Wvh
  dst[3 * WSZ + i] = f2bf(Wqkv[r * 1536 + c]);          // Wqh
  dst[4 * WSZ + i] = f2bf(Wq[i]);
  dst[5 * WSZ + i] = f2bf(Wkv[0 * 512 * 1024 + r * 1024 + c]);
  dst[6 * WSZ + i] = f2bf(Wkv[1 * 512 * 1024 + r * 1024 + c]);
  dst[7 * WSZ + i] = f2bf(Wkv[2 * 512 * 1024 + r * 1024 + c]);
  dst[8 * WSZ + i] = f2bf(Wao[i]);
  dst[9 * WSZ + i] = f2bf(Wo[i]);
}

// ---------------------------------------------------------------------------
// WMMA GEMM: C[M,N] = act(A @ B + bias). Block tile 64x128, BK=32.
// 8 waves (2x4); each wave computes 32x32 = 2x2 WMMA frags (4 wmma/k-step,
// A/B fragments each reused twice). AF32: A is f32, converted while staging.
// ---------------------------------------------------------------------------
#define TILE_P 40

template <int AF32>
__global__ __launch_bounds__(256)
void wmma_gemm_bias(const void* __restrict__ Aany,
                    const unsigned short* __restrict__ Bm,
                    const float* __restrict__ bias,
                    unsigned short* __restrict__ Cbf,
                    float* __restrict__ Cf,
                    int M, int N, int K, int act) {
  __shared__ unsigned short As[64 * TILE_P];
  __shared__ unsigned short Bs[128 * TILE_P];   // transposed: [n][k]

  const int tid  = threadIdx.x;
  const int lane = tid & 31;
  const int wave = tid >> 5;
  const int wr   = wave >> 2;        // 0..1
  const int wc   = wave & 3;         // 0..3
  const int bm0  = blockIdx.y * 64;
  const int bn0  = blockIdx.x * 128;
  const int l16  = lane & 15;
  const int hi8  = (lane >> 4) << 3;
  const int hi16 = (lane >> 4) << 4;

  v8f acc[2][2];
#pragma unroll
  for (int i = 0; i < 2; ++i)
#pragma unroll
    for (int j = 0; j < 2; ++j) acc[i][j] = v8f_zero();

  const int aidx = tid << 3;                   // A: 8 halves/thread
  const int ar = aidx >> 5, ac = aidx & 31;    // A tile 64x32
  const int bidx = tid << 4;                   // B: 16 halves/thread
  const int br = bidx >> 7, bc = bidx & 127;   // B tile 32x128

  for (int k0 = 0; k0 < K; k0 += 32) {
    // ---- stage A tile (64x32) ----
    if (AF32) {
      const float* Af = (const float*)Aany + (size_t)(bm0 + ar) * K + (k0 + ac);
      float4 f0 = *(const float4*)(Af);
      float4 f1 = *(const float4*)(Af + 4);
      unsigned short* d = &As[ar * TILE_P + ac];
      d[0] = f2bf(f0.x); d[1] = f2bf(f0.y); d[2] = f2bf(f0.z); d[3] = f2bf(f0.w);
      d[4] = f2bf(f1.x); d[5] = f2bf(f1.y); d[6] = f2bf(f1.z); d[7] = f2bf(f1.w);
    } else {
      const unsigned short* Ah =
          (const unsigned short*)Aany + (size_t)(bm0 + ar) * K + (k0 + ac);
      *(uint4*)&As[ar * TILE_P + ac] = *(const uint4*)Ah;
    }
    // ---- stage B tile (32x128) transposed into Bs[n][k] ----
    {
      const unsigned short* Bh = Bm + (size_t)(k0 + br) * N + (bn0 + bc);
      uint4 v0 = *(const uint4*)Bh;
      uint4 v1 = *(const uint4*)(Bh + 8);
      unsigned short tmp[16];
      *(uint4*)tmp = v0;
      *(uint4*)(tmp + 8) = v1;
#pragma unroll
      for (int j = 0; j < 16; ++j) Bs[(bc + j) * TILE_P + br] = tmp[j];
    }
    if (k0 + 32 < K) {   // -> global_prefetch_b8
      if (AF32)
        __builtin_prefetch((const float*)Aany + (size_t)(bm0 + ar) * K + (k0 + 32 + ac), 0, 1);
      else
        __builtin_prefetch((const unsigned short*)Aany + (size_t)(bm0 + ar) * K + (k0 + 32 + ac), 0, 1);
      __builtin_prefetch(Bm + (size_t)(k0 + 32 + br) * N + (bn0 + bc), 0, 1);
    }
    __syncthreads();

    // ---- A (16x32) / B (32x16) fragments, contiguous k = 16*(lane/16) + j ----
    Frag16 a0, a1, b0, b1;
    const int m0 = (wr * 32 + l16) * TILE_P;
    const int m1 = m0 + 16 * TILE_P;
    const int n0 = (wc * 32 + l16) * TILE_P + hi16;
    const int n1 = n0 + 16 * TILE_P;
#pragma unroll
    for (int j = 0; j < 16; ++j) {
      int ka = hi16 + j;
      a0.s[j] = As[m0 + ka];
      a1.s[j] = As[m1 + ka];
      b0.s[j] = Bs[n0 + j];
      b1.s[j] = Bs[n1 + j];
    }
    acc[0][0] = __builtin_amdgcn_wmma_f32_16x16x32_bf16(false, a0.b, false, b0.b,
                                                        (short)0, acc[0][0], false, false);
    acc[0][1] = __builtin_amdgcn_wmma_f32_16x16x32_bf16(false, a0.b, false, b1.b,
                                                        (short)0, acc[0][1], false, false);
    acc[1][0] = __builtin_amdgcn_wmma_f32_16x16x32_bf16(false, a1.b, false, b0.b,
                                                        (short)0, acc[1][0], false, false);
    acc[1][1] = __builtin_amdgcn_wmma_f32_16x16x32_bf16(false, a1.b, false, b1.b,
                                                        (short)0, acc[1][1], false, false);
    __syncthreads();
  }

  const int ng0 = bn0 + wc * 32 + l16;
  const int ng1 = ng0 + 16;
  const float bv0 = bias ? bias[ng0] : 0.0f;
  const float bv1 = bias ? bias[ng1] : 0.0f;
#pragma unroll
  for (int rt = 0; rt < 2; ++rt)
#pragma unroll
    for (int ct = 0; ct < 2; ++ct) {
      const int ng = ct ? ng1 : ng0;
      const float bvv = ct ? bv1 : bv0;
#pragma unroll
      for (int i = 0; i < 8; ++i) {
        int mg = bm0 + wr * 32 + rt * 16 + i + hi8;
        if (mg < M) {
          float v = acc[rt][ct][i] + bvv;
          if (act == 1) v = tanhf(v);
          size_t o = (size_t)mg * N + ng;
          if (Cbf) Cbf[o] = f2bf(v);
          if (Cf)  Cf[o]  = v;
        }
      }
    }
}

// ---------------------------------------------------------------------------
// Spike scoring, top-32 selection, gaussian window pooling -> Z (bf16), gates.
// One block per (k,b).
// ---------------------------------------------------------------------------
__global__ __launch_bounds__(256)
void spike_z_kernel(const float* __restrict__ A,
                    const int* __restrict__ spikes,
                    const float* __restrict__ h_ctc,
                    unsigned short* __restrict__ Zbf,
                    float* __restrict__ gk,
                    int B, int T, int S0) {
  const int b = blockIdx.x % B;
  const int k = blockIdx.x / B;
  const int tid = threadIdx.x;
  const float* Ak = A + ((size_t)k * B + b) * T;
  const int* sp = spikes + ((size_t)k * B + b) * S0;
  const float* Hk = h_ctc + (((size_t)k * B + b) * T) * 512;

  __shared__ float sc[128];
  __shared__ int sel[32];
  __shared__ unsigned char used[128];

  for (int s = tid; s < S0; s += blockDim.x) {
    int t = sp[s];
    float v = -1.0e9f;
    if (t >= 0 && t < T) {
      float sum = 0.0f; int cnt = 0;
      for (int o = -8; o <= 8; ++o) {
        int tc = t + o;
        if (tc >= 0 && tc < T) { sum += Ak[tc]; ++cnt; }
      }
      v = sum / (float)(cnt > 0 ? cnt : 1);
    }
    sc[s] = v;
    used[s] = 0;
  }
  __syncthreads();
  if (tid == 0) {  // stable descending top-32 (matches jax.lax.top_k ordering)
    for (int i = 0; i < 32; ++i) {
      int best = 0; float bvv = -3.0e38f;
      for (int s = 0; s < S0; ++s)
        if (!used[s] && sc[s] > bvv) { bvv = sc[s]; best = s; }
      used[best] = 1;
      sel[i] = best;
    }
  }
  __syncthreads();

  for (int s = 0; s < 32; ++s) {
    const int t = sp[sel[s]];
    const bool okt = (t >= 0 && t < T);
    float w[17]; float wsum = 0.0f;
#pragma unroll
    for (int o = 0; o < 17; ++o) {
      int tt = t + o - 8;
      bool v = okt && (tt >= 0) && (tt < T);
      int tc = tt < 0 ? 0 : (tt > T - 1 ? T - 1 : tt);
      float off = (float)(o - 8) * 0.25f;               // /SIGMA=4
      float g = __expf(-0.5f * off * off);
      float wv = v ? g * Ak[tc] : 0.0f;
      w[o] = wv; wsum += wv;
    }
    const float inv = 1.0f / (wsum + 1e-6f);
    const size_t zrow = (((size_t)k * B + b) * 32 + s) * 512;
    for (int d = tid; d < 512; d += blockDim.x) {
      float acc = 0.0f;
#pragma unroll
      for (int o = 0; o < 17; ++o) {
        int tt = t + o - 8;
        int tc = tt < 0 ? 0 : (tt > T - 1 ? T - 1 : tt);
        acc += w[o] * Hk[(size_t)tc * 512 + d];
      }
      Zbf[zrow + d] = f2bf(acc * inv);
    }
    if (tid == 0) {
      float conf = 0.0f;
      if (okt) {
        float sum = 0.0f; int cnt = 0;
        for (int o = -8; o <= 8; ++o) {
          int tc = t + o;
          if (tc >= 0 && tc < T) { sum += Ak[tc]; ++cnt; }
        }
        conf = sum / (float)(cnt > 0 ? cnt : 1);
      }
      float vm = okt ? 1.0f : 0.0f;
      gk[((size_t)k * B + b) * 32 + s] = vm / (1.0f + __expf(-2.0f * conf));
    }
  }
}

// ---------------------------------------------------------------------------
// Flash attention: 1 wave per (k,b,h). 32 queries x T keys, hd=64.
// K/V tiles (32x64 bf16) streamed by TDM into LDS (padded pitch 72),
// online softmax, QK^T and PV via WMMA bf16.
// ---------------------------------------------------------------------------
#define QP 72
#define PP 40

__global__ __launch_bounds__(32)
void flash_attn_kernel(const unsigned short* __restrict__ Qbf,   // (3*B*32,512)
                       const unsigned short* __restrict__ Kmem,  // (B*T,512)
                       const unsigned short* __restrict__ Vmem,  // (B*T,512)
                       unsigned short* __restrict__ Ctx,         // (3*B*32,512)
                       int B, int T, int H) {
  __shared__ unsigned short Qs[32 * QP];
  __shared__ unsigned short Ks[32 * QP];
  __shared__ unsigned short Vs[32 * QP];
  __shared__ unsigned short Ps[32 * PP];

  const int blk = blockIdx.x;
  const int h = blk % H;
  const int b = (blk / H) % B;
  const int k = blk / (H * B);
  const int lane = threadIdx.x & 31;
  const int l16 = lane & 15;
  const int hi8 = (lane >> 4) << 3;
  const int hi16 = (lane >> 4) << 4;

  const size_t qbase = (((size_t)(k * B + b)) * 32) * 512 + (size_t)h * 64;

#pragma unroll
  for (int j = 0; j < 64; j += 8)
    *(uint4*)&Qs[lane * QP + j] = *(const uint4*)&Qbf[qbase + (size_t)lane * 512 + j];
  __syncthreads();

  Frag16 aq[2][2];   // [rowtile][kstep]: A = Q (16x32 each)
#pragma unroll
  for (int rt = 0; rt < 2; ++rt)
#pragma unroll
    for (int ks = 0; ks < 2; ++ks)
#pragma unroll
      for (int j = 0; j < 16; ++j)
        aq[rt][ks].s[j] =
            Qs[(rt * 16 + l16) * QP + ks * 32 + hi16 + j];

  float mrow[2][8], lrow[2][8];
  v8f O[2][4];
#pragma unroll
  for (int rt = 0; rt < 2; ++rt) {
#pragma unroll
    for (int i = 0; i < 8; ++i) { mrow[rt][i] = -3.0e38f; lrow[rt][i] = 0.0f; }
#pragma unroll
    for (int nt = 0; nt < 4; ++nt) O[rt][nt] = v8f_zero();
  }

  const float scale = 0.125f;   // 1/sqrt(64)
  const size_t kvbase = ((size_t)b * T) * 512 + (size_t)h * 64;
#if HAVE_TDM
  const unsigned ksOff = (unsigned)(size_t)(void*)Ks;
  const unsigned vsOff = (unsigned)(size_t)(void*)Vs;
#endif

  for (int t0 = 0; t0 < T; t0 += 32) {
#if HAVE_TDM
    // TDM: one DMA per 32x64 bf16 tile, LDS pad -> pitch 72 halves
    tdm_load_2d_bf16(ksOff, &Kmem[kvbase + (size_t)t0 * 512], 64, 32, 512);
    tdm_load_2d_bf16(vsOff, &Vmem[kvbase + (size_t)t0 * 512], 64, 32, 512);
    __builtin_amdgcn_s_wait_tensorcnt(0);
#else
    const size_t g = kvbase + (size_t)(t0 + lane) * 512;
#pragma unroll
    for (int j = 0; j < 64; j += 8) {
      *(uint4*)&Ks[lane * QP + j] = *(const uint4*)&Kmem[g + j];
      *(uint4*)&Vs[lane * QP + j] = *(const uint4*)&Vmem[g + j];
    }
    if (t0 + 32 < T) {
      __builtin_prefetch(&Kmem[g + 32 * 512], 0, 1);
      __builtin_prefetch(&Vmem[g + 32 * 512], 0, 1);
    }
#endif
    __syncthreads();

    // S = Q K^T (32x32 chunk) : B-fragment lane l -> key = ct*16+l16
    v8f S[2][2];
#pragma unroll
    for (int rt = 0; rt < 2; ++rt)
#pragma unroll
      for (int ct = 0; ct < 2; ++ct) {
        v8f s = v8f_zero();
#pragma unroll
        for (int ks = 0; ks < 2; ++ks) {
          Frag16 bk;
#pragma unroll
          for (int j = 0; j < 16; ++j)
            bk.s[j] = Ks[(ct * 16 + l16) * QP + ks * 32 + hi16 + j];
          s = __builtin_amdgcn_wmma_f32_16x16x32_bf16(false, aq[rt][ks].b, false,
                                                      bk.b, (short)0, s, false, false);
        }
        S[rt][ct] = s;
      }

    // online softmax: rows live in 16-lane groups -> shfl_xor reductions
    float p0a[2][8], p1a[2][8];
#pragma unroll
    for (int rt = 0; rt < 2; ++rt) {
#pragma unroll
      for (int i = 0; i < 8; ++i) {
        float v0 = S[rt][0][i] * scale;
        float v1 = S[rt][1][i] * scale;
        float mx = fmaxf(v0, v1);
#pragma unroll
        for (int d = 1; d < 16; d <<= 1) mx = fmaxf(mx, __shfl_xor(mx, d, 32));
        float mn = fmaxf(mrow[rt][i], mx);
        float p0 = __expf(v0 - mn);
        float p1 = __expf(v1 - mn);
        float rs = p0 + p1;
#pragma unroll
        for (int d = 1; d < 16; d <<= 1) rs += __shfl_xor(rs, d, 32);
        float corr = __expf(mrow[rt][i] - mn);
        lrow[rt][i] = lrow[rt][i] * corr + rs;
        mrow[rt][i] = mn;
#pragma unroll
        for (int nt = 0; nt < 4; ++nt) O[rt][nt][i] *= corr;
        p0a[rt][i] = p0; p1a[rt][i] = p1;
      }
    }

    // transpose P (D-layout) into LDS as bf16 for A-fragment reuse
#pragma unroll
    for (int rt = 0; rt < 2; ++rt)
#pragma unroll
      for (int i = 0; i < 8; ++i) {
        int row = rt * 16 + i + hi8;
        Ps[row * PP + l16]      = f2bf(p0a[rt][i]);
        Ps[row * PP + 16 + l16] = f2bf(p1a[rt][i]);
      }
    __syncthreads();

    // O += P V  : A = P (16x32), B = V chunk (32 keys x 16 dims)
#pragma unroll
    for (int rt = 0; rt < 2; ++rt) {
      Frag16 ap;
#pragma unroll
      for (int j = 0; j < 16; ++j)
        ap.s[j] = Ps[(rt * 16 + l16) * PP + hi16 + j];
#pragma unroll
      for (int nt = 0; nt < 4; ++nt) {
        Frag16 bv_;
#pragma unroll
        for (int j = 0; j < 16; ++j)
          bv_.s[j] = Vs[(hi16 + j) * QP + nt * 16 + l16];
        O[rt][nt] = __builtin_amdgcn_wmma_f32_16x16x32_bf16(
            false, ap.b, false, bv_.b, (short)0, O[rt][nt], false, false);
      }
    }
    __syncthreads();   // ds reads done before next TDM/stage overwrites
  }

  // epilogue: ctx = O / l
#pragma unroll
  for (int rt = 0; rt < 2; ++rt)
#pragma unroll
    for (int nt = 0; nt < 4; ++nt)
#pragma unroll
      for (int i = 0; i < 8; ++i) {
        int row = rt * 16 + i + hi8;
        float v = O[rt][nt][i] / lrow[rt][i];
        Ctx[qbase + (size_t)row * 512 + nt * 16 + l16] = f2bf(v);
      }
}

// ---------------------------------------------------------------------------
// Final gating + track concatenation: out[b, k*32+s, d] = t2[row,d] * gk[row]
// ---------------------------------------------------------------------------
__global__ void gate_out_kernel(const float* __restrict__ t2,
                                const float* __restrict__ gk,
                                float* __restrict__ out, int B) {
  int i = blockIdx.x * blockDim.x + threadIdx.x;
  const int total = 3 * B * 32 * 512;
  if (i >= total) return;
  int d = i & 511;
  int row = i >> 9;          // (k*B + b)*32 + s
  int s = row & 31;
  int kb = row >> 5;
  int b = kb % B;
  int k = kb / B;
  out[(((size_t)b * 96) + (size_t)k * 32 + s) * 512 + d] = t2[i] * gk[row];
}

// ---------------------------------------------------------------------------
extern "C" void kernel_launch(void* const* d_in, const int* in_sizes, int n_in,
                              void* d_out, int out_size, void* d_ws, size_t ws_size,
                              hipStream_t stream) {
  (void)in_sizes; (void)n_in; (void)out_size; (void)ws_size;
  const float* proj   = (const float*)d_in[0];
  const float* h_ctc  = (const float*)d_in[1];
  const float* A      = (const float*)d_in[2];
  const int*   spikes = (const int*)d_in[3];
  const float* W_mem  = (const float*)d_in[4];
  const float* b_mem  = (const float*)d_in[5];
  const float* W_kv   = (const float*)d_in[6];
  const float* b_kv   = (const float*)d_in[7];
  const float* W_q    = (const float*)d_in[8];
  const float* b_q    = (const float*)d_in[9];
  const float* W_qkv  = (const float*)d_in[10];
  const float* b_qkv  = (const float*)d_in[11];
  const float* W_ao   = (const float*)d_in[12];
  const float* b_ao   = (const float*)d_in[13];
  const float* W_o    = (const float*)d_in[14];
  const float* b_o    = (const float*)d_in[15];

  constexpr int Kspk = 3, B = 8, T = 8192, S0 = 128, D = 512, H = 8;
  const size_t MT  = (size_t)B * T;      // 65536 rows
  const size_t WSZ = (size_t)D * D;      // 262144

  char* ws = (char*)d_ws;
  size_t off = 0;
  auto take = [&](size_t bytes) {
    char* p = ws + off;
    off += (bytes + 255) & ~(size_t)255;
    return p;
  };

  unsigned short* Wb   = (unsigned short*)take(10 * WSZ * 2);
  unsigned short* Mbf  = (unsigned short*)take(MT * D * 2);
  unsigned short* Kbf  = (unsigned short*)take(MT * D * 2);
  unsigned short* Vbf  = (unsigned short*)take(MT * D * 2);
  unsigned short* Zbf  = (unsigned short*)take((size_t)768 * D * 2);
  unsigned short* KSbf = (unsigned short*)take((size_t)768 * D * 2);
  unsigned short* QKbf = (unsigned short*)take((size_t)768 * D * 2);
  unsigned short* Qbf  = (unsigned short*)take((size_t)768 * D * 2);
  unsigned short* CXbf = (unsigned short*)take((size_t)768 * D * 2);
  unsigned short* T1bf = (unsigned short*)take((size_t)768 * D * 2);
  float*          T2f  = (float*)take((size_t)768 * D * 4);
  float*          Gk   = (float*)take((size_t)768 * 4);

  pack_weights_kernel<<<(int)((WSZ + 255) / 256), 256, 0, stream>>>(
      W_mem, W_qkv, W_q, W_kv, W_ao, W_o, Wb);

  dim3 blk(256);
  dim3 gBig(D / 128, (int)(MT / 64));      // (4, 1024)
  // M = proj @ W_mem + b_mem   (A is f32, converted while staging)
  wmma_gemm_bias<1><<<gBig, blk, 0, stream>>>(proj, Wb + 0 * WSZ, b_mem,
                                              Mbf, nullptr, (int)MT, D, D, 0);
  // k_mem / v_mem
  wmma_gemm_bias<0><<<gBig, blk, 0, stream>>>(Mbf, Wb + 1 * WSZ, b_qkv + 512,
                                              Kbf, nullptr, (int)MT, D, D, 0);
  wmma_gemm_bias<0><<<gBig, blk, 0, stream>>>(Mbf, Wb + 2 * WSZ, b_qkv + 1024,
                                              Vbf, nullptr, (int)MT, D, D, 0);

  spike_z_kernel<<<Kspk * B, 256, 0, stream>>>(A, spikes, h_ctc, Zbf, Gk, B, T, S0);

  dim3 gK(D / 128, 256 / 64);              // per-speaker K_seed (4,4)
  for (int k = 0; k < Kspk; ++k)
    wmma_gemm_bias<0><<<gK, blk, 0, stream>>>(
        Zbf + (size_t)k * 256 * D, Wb + (5 + k) * WSZ, b_kv + k * 1024,
        KSbf + (size_t)k * 256 * D, nullptr, 256, D, D, 0);

  dim3 gAll(D / 128, 768 / 64);            // (4,12)
  wmma_gemm_bias<0><<<gAll, blk, 0, stream>>>(KSbf, Wb + 4 * WSZ, b_q,
                                              QKbf, nullptr, 768, D, D, 1);  // tanh
  wmma_gemm_bias<0><<<gAll, blk, 0, stream>>>(QKbf, Wb + 3 * WSZ, b_qkv + 0,
                                              Qbf, nullptr, 768, D, D, 0);

  flash_attn_kernel<<<Kspk * B * H, 32, 0, stream>>>(Qbf, Kbf, Vbf, CXbf, B, T, H);

  wmma_gemm_bias<0><<<gAll, blk, 0, stream>>>(CXbf, Wb + 8 * WSZ, b_ao,
                                              T1bf, nullptr, 768, D, D, 0);
  wmma_gemm_bias<0><<<gAll, blk, 0, stream>>>(T1bf, Wb + 9 * WSZ, b_o,
                                              nullptr, T2f, 768, D, D, 0);

  gate_out_kernel<<<(768 * D + 255) / 256, 256, 0, stream>>>(T2f, Gk,
                                                             (float*)d_out, B);
}
